// GeneralizedGaussianMomentDescriptor_25417616458488
// MI455X (gfx1250) — matrix operand order repaired and checked by reading, not compiled, not run on hardware
//
#include <hip/hip_runtime.h>
#include <math.h>

#define N_ATOMS   10000
#define N_EDGES   400000
#define N_RADIAL  8
#define N_BASIS   16
#define N_SPECIES 10
#define NPAIR     (N_SPECIES * N_SPECIES)
#define R_MAX_F   6.0f
#define BETA_F    4.0f
#define NMOMU     20            // unique monomials: 1 + 3 + 6 + 10
#define MOM_STRIDE (N_RADIAL * NMOMU)   // 160 floats per atom
#define OUT_STRIDE 2248         // 8 + 3*64 + 4*512
#define MAX_SORT  401504        // 400000 + 100*15 padding, rounded up
#define CTRL_N    308           // binCount[100], binStart[101], cursor[100], misc
#define PI_F      3.14159265358979f
#define W_ELEMS   (NPAIR * N_RADIAL * N_BASIS)   // 12800 f32 = 51.2 KB

typedef __attribute__((ext_vector_type(16))) _Float16 v16h;
typedef __attribute__((ext_vector_type(8)))  float    v8f;
typedef __attribute__((ext_vector_type(4)))  unsigned int u32x4;
typedef __attribute__((ext_vector_type(4)))  int      i32x4;
typedef __attribute__((ext_vector_type(8)))  int      i32x8;

// Symmetric-component index tables (fold at compile time under full unroll).
// unique M2 order: (00),(01),(02),(11),(12),(22)
__device__ constexpr int U2TAB[9] = {0,1,2, 1,3,4, 2,4,5};
// unique M3 order: (000),(001),(002),(011),(012),(022),(111),(112),(122),(222)
__device__ constexpr int U3TAB[27] = {
  0,1,2, 1,3,4, 2,4,5,
  1,3,4, 3,6,7, 4,7,8,
  2,4,5, 4,7,8, 5,8,9 };

#if __has_builtin(__builtin_amdgcn_tensor_load_to_lds)
#define HAVE_TDM 1
#else
#define HAVE_TDM 0
#endif

// ---------------- K0: zero / sentinel init of workspace ----------------
__global__ void k_init(float* __restrict__ mom, int* __restrict__ sorted,
                       int* __restrict__ ctrl) {
  int i = blockIdx.x * 256 + threadIdx.x;
  if (i < N_ATOMS * MOM_STRIDE) mom[i] = 0.0f;
  if (i < MAX_SORT)             sorted[i] = -1;
  if (i < CTRL_N)               ctrl[i] = 0;
}

// ---------------- K1: histogram edges by species pair ----------------
__global__ void k_hist(const int* __restrict__ Z, const int* __restrict__ nbr,
                       int* __restrict__ ctrl) {
  int e = blockIdx.x * 256 + threadIdx.x;
  if (e >= N_EDGES) return;
  int ii = nbr[e], jj = nbr[N_EDGES + e];
  if (ii == jj) return;                       // masked edges never enter bins
  int p = Z[ii] * N_SPECIES + Z[jj];
  atomicAdd(&ctrl[p], 1);
}

// ---------------- K2: serial 100-bin scan, pad bins to multiples of 16 ----
__global__ void k_scan(int* __restrict__ ctrl) {
  if (threadIdx.x == 0 && blockIdx.x == 0) {
    int acc = 0;
    for (int p = 0; p < NPAIR; ++p) {
      ctrl[100 + p] = acc;                    // binStart (padded)
      ctrl[201 + p] = acc;                    // cursor
      acc += (ctrl[p] + 15) & ~15;
    }
    ctrl[100 + NPAIR] = acc;                  // paddedTotal
    ctrl[302] = acc >> 4;                     // numTiles
  }
}

// ---------------- K3: scatter edge ids into padded, pair-sorted order -----
__global__ void k_scatter(const int* __restrict__ Z, const int* __restrict__ nbr,
                          int* __restrict__ ctrl, int* __restrict__ sorted) {
  int e = blockIdx.x * 256 + threadIdx.x;
  if (e >= N_EDGES) return;
  int ii = nbr[e], jj = nbr[N_EDGES + e];
  if (ii == jj) return;
  int p = Z[ii] * N_SPECIES + Z[jj];
  int pos = atomicAdd(&ctrl[201 + p], 1);
  sorted[pos] = e;
}

// ---------------- K4: edge kernel — TDM W-stage, WMMA radial GEMM, scatter -
// One wave handles one tile of 16 same-species-pair edges.
// radial(16x8) = X(16x16, f16) * Wp^T(16x8, f16) via v_wmma_f32_16x16x32_f16.
__global__ void __launch_bounds__(256)
k_edges(const float* __restrict__ drv, const int* __restrict__ Z,
        const int* __restrict__ nbr, const float* __restrict__ W,
        const int* __restrict__ sorted, const int* __restrict__ ctrl,
        float* __restrict__ mom) {
  __shared__ float Wlds[W_ELEMS];             // 51.2 KB f32 W table

#if HAVE_TDM
  // Tensor Data Mover: one 1-row descriptor moves all of W (51200 B) to LDS.
  if (threadIdx.x < 32) {                     // wave 0 issues the DMA
    unsigned lds_off = (unsigned)(unsigned long long)(void*)Wlds; // LDS byte off
    unsigned long long ga = (unsigned long long)W;
    u32x4 g0;
    g0[0] = 1u;                               // count=1, user descriptor
    g0[1] = lds_off;                          // lds_addr [63:32]
    g0[2] = (unsigned)(ga & 0xffffffffu);     // global_addr [95:64]
    g0[3] = (unsigned)((ga >> 32) & 0x1ffffffu) | (2u << 30); // addr hi | type=2
    i32x8 g1;
    g1[0] = 0x20000;                          // data_size=2 (4 B elements)
    g1[1] = (int)((unsigned)W_ELEMS << 16);   // tensor_dim0[15:0] @ [63:48]
    g1[2] = (int)(1u << 16);                  // tensor_dim0 hi=0 | tensor_dim1=1
    g1[3] = (int)((unsigned)W_ELEMS << 16);   // tile_dim0 @ [127:112]
    g1[4] = 1;                                // tile_dim1=1, tile_dim2=0
    g1[5] = W_ELEMS;                          // tensor_dim0_stride lo32
    g1[6] = 0;
    g1[7] = 0;
    i32x4 z4 = {0, 0, 0, 0};
#if __clang_major__ >= 23
    i32x8 z8 = {0, 0, 0, 0, 0, 0, 0, 0};
    __builtin_amdgcn_tensor_load_to_lds(g0, g1, z4, z4, z8, 0);
#else
    __builtin_amdgcn_tensor_load_to_lds(g0, g1, z4, z4, 0);
#endif
    __builtin_amdgcn_s_wait_tensorcnt(0);
  }
  __syncthreads();
#else
  for (int t = threadIdx.x; t < W_ELEMS; t += 256) Wlds[t] = W[t];
  __syncthreads();
#endif

  const int numTiles = ctrl[302];
  const int tile = blockIdx.x * 8 + (threadIdx.x >> 5);
  if (tile >= numTiles) return;               // wave-uniform; no barriers below

  const int lane = threadIdx.x & 31;
  const int el = lane & 15;                   // edge-slot (and radial col N)
  const int h  = lane >> 4;                   // wave half

  __builtin_prefetch(&sorted[(tile + 8) * 16], 0, 0);  // global_prefetch

  const int e  = sorted[tile * 16 + el];
  const int e0 = sorted[tile * 16];           // guaranteed >= 0 (padding < 16)
  const int p  = Z[nbr[e0]] * N_SPECIES + Z[nbr[N_EDGES + e0]];

  float dnx = 0.f, dny = 0.f, dnz = 0.f;
  int   jv  = -1;
  float X[N_BASIS];
#pragma unroll
  for (int b = 0; b < N_BASIS; ++b) X[b] = 0.f;

  if (e >= 0) {
    float x = drv[3 * e], y = drv[3 * e + 1], z = drv[3 * e + 2];
    float dr = sqrtf(x * x + y * y + z * z);
    float inv = 1.0f / (dr + 1e-5f);
    dnx = x * inv; dny = y * inv; dnz = z * inv;
    float cut = (dr < R_MAX_F) ? 0.5f * (__cosf(PI_F * dr / R_MAX_F) + 1.0f) : 0.0f;
    jv = nbr[N_EDGES + e];
    const float dmu = R_MAX_F / (float)(N_BASIS - 1);
#pragma unroll
    for (int b = 0; b < N_BASIS; ++b) {
      float d = dr - dmu * (float)b;
      X[b] = __expf(-BETA_F * d * d) * cut;
    }
  }

  // A (16x32 f16): lane holds K = 8h..8h+7 in elements 0..7; K>=16 zero-padded
  v16h a;
#pragma unroll
  for (int q = 0; q < 8; ++q)  a[q] = (_Float16)X[8 * h + q];
#pragma unroll
  for (int q = 8; q < 16; ++q) a[q] = (_Float16)0.0f;

  // B (32x16 f16): lane holds K = 8h..8h+7 in elements 0..7 for column N=el; K>=16 zero
  v16h bm;
  const float* Wp = &Wlds[p * (N_RADIAL * N_BASIS)];
#pragma unroll
  for (int q = 0; q < 16; ++q)
    bm[q] = (q < 8 && el < N_RADIAL) ? (_Float16)Wp[el * N_BASIS + 8 * h + q]
                                      : (_Float16)0.0f;

  v8f c = {};
  c = __builtin_amdgcn_wmma_f32_16x16x32_f16(false, a, false, bm,
                                             (short)0, c, false, false);
  // c[v] on lane l = radial[edge slot v + 8h][r = el]  (valid for el < 8)

  // Scatter-add the 20 UNIQUE symmetric moment monomials (halves atomics).
#pragma unroll
  for (int v = 0; v < 8; ++v) {
    int s = 8 * h + v;                        // edge slot this c[v] refers to
    int   js = __shfl(jv,  s);
    float sx = __shfl(dnx, s);
    float sy = __shfl(dny, s);
    float sz = __shfl(dnz, s);
    if (el < N_RADIAL && js >= 0) {
      float rad = c[v];
      float* base = mom + (size_t)js * MOM_STRIDE + el * NMOMU;
      float d3[3] = {sx, sy, sz};
      atomicAdd(base + 0, rad);
#pragma unroll
      for (int i = 0; i < 3; ++i) {
        float r1 = rad * d3[i];
        atomicAdd(base + 1 + i, r1);
#pragma unroll
        for (int j = i; j < 3; ++j) {
          float r2 = r1 * d3[j];
          atomicAdd(base + 4 + U2TAB[i * 3 + j], r2);
#pragma unroll
          for (int k = j; k < 3; ++k)
            atomicAdd(base + 10 + U3TAB[i * 9 + j * 3 + k], r2 * d3[k]);
        }
      }
    }
  }
}

// ---------------- K5: per-atom tensor contractions ----------------
// Unique-moment layout per (atom, r):
//   [0]=M0, [1+i]=M1, [4+u2]=M2 unique, [10+u3]=M3 unique.
// Accessors expand symmetric components via compile-time tables.
__global__ void __launch_bounds__(256)
k_atoms(const float* __restrict__ mom, float* __restrict__ out) {
  __shared__ float Ms[8 * MOM_STRIDE];        // 5.1 KB: 8 atoms per block
  const int w = threadIdx.x >> 5, lane = threadIdx.x & 31;
  const int a = blockIdx.x * 8 + w;
  const float* src = mom + (size_t)a * MOM_STRIDE;
  float* Ma = &Ms[w * MOM_STRIDE];
  for (int q = lane; q < MOM_STRIDE; q += 32) Ma[q] = src[q];
  __syncthreads();

  float* O = out + (size_t)a * OUT_STRIDE;
#define M0_(r)       Ma[(r)*NMOMU]
#define M1_(r,i)     Ma[(r)*NMOMU + 1 + (i)]
#define M2F(r,m)     Ma[(r)*NMOMU + 4 + U2TAB[(m)]]
#define M3F(r,m)     Ma[(r)*NMOMU + 10 + U3TAB[(m)]]

  if (lane < 8) O[lane] = M0_(lane);

  // Pair contractions (1,1),(2,2),(3,3): 64 outputs each, 2 per lane
#pragma unroll
  for (int u = 0; u < 2; ++u) {
    int o = lane * 2 + u, r = o >> 3, s = o & 7;
    float s1 = 0.f, s2 = 0.f, s3 = 0.f;
#pragma unroll
    for (int i = 0; i < 3; ++i)  s1 += M1_(r, i) * M1_(s, i);
#pragma unroll
    for (int m = 0; m < 9; ++m)  s2 += M2F(r, m) * M2F(s, m);
#pragma unroll
    for (int m = 0; m < 27; ++m) s3 += M3F(r, m) * M3F(s, m);
    O[8 + o] = s1; O[72 + o] = s2; O[136 + o] = s3;
  }

  // Triple contractions: o = lane*16 + half*8 + t; r = lane>>2 fixed per lane
  const int r = lane >> 2;
  float M2r[9], M3r[27];
#pragma unroll
  for (int m = 0; m < 9; ++m)  M2r[m] = M2F(r, m);
#pragma unroll
  for (int m = 0; m < 27; ++m) M3r[m] = M3F(r, m);

  for (int half = 0; half < 2; ++half) {
    const int s = (lane * 2 + half) & 7;
    float M1s[3], M2s[9], M3s[27];
#pragma unroll
    for (int i = 0; i < 3; ++i)  M1s[i] = M1_(s, i);
#pragma unroll
    for (int m = 0; m < 9; ++m)  M2s[m] = M2F(s, m);
#pragma unroll
    for (int m = 0; m < 27; ++m) M3s[m] = M3F(s, m);

    // (3,2,1) factor: A321[k] = sum_ij M3r[ijk]*M2s[ij]
    float A321[3] = {0.f, 0.f, 0.f};
    // (3,3,2) factor: D[k,l]  = sum_ij M3r[ijk]*M3s[ijl]
    float D[9];
#pragma unroll
    for (int m = 0; m < 9; ++m) D[m] = 0.f;
#pragma unroll
    for (int i = 0; i < 3; ++i)
#pragma unroll
      for (int j = 0; j < 3; ++j) {
        float m2 = M2s[i * 3 + j];
#pragma unroll
        for (int k = 0; k < 3; ++k) {
          float m3 = M3r[i * 9 + j * 3 + k];
          A321[k] += m3 * m2;
#pragma unroll
          for (int l = 0; l < 3; ++l)
            D[k * 3 + l] += m3 * M3s[i * 9 + j * 3 + l];
        }
      }

    for (int t = 0; t < 8; ++t) {
      const int o = lane * 16 + half * 8 + t;
      float M1t[3] = {M1_(t, 0), M1_(t, 1), M1_(t, 2)};
      float M2t[9];
#pragma unroll
      for (int m = 0; m < 9; ++m) M2t[m] = M2F(t, m);

      float c211 = 0.f, c222 = 0.f, c332 = 0.f;
#pragma unroll
      for (int i = 0; i < 3; ++i)
#pragma unroll
        for (int j = 0; j < 3; ++j) {
          c211 += M2r[i * 3 + j] * M1s[i] * M1t[j];
#pragma unroll
          for (int k = 0; k < 3; ++k)
            c222 += M2r[i * 3 + j] * M2s[i * 3 + k] * M2t[j * 3 + k];
        }
      float c321 = A321[0] * M1t[0] + A321[1] * M1t[1] + A321[2] * M1t[2];
#pragma unroll
      for (int m = 0; m < 9; ++m) c332 += D[m] * M2t[m];

      O[200  + o] = c211;   // (2,1,1)
      O[712  + o] = c321;   // (3,2,1)
      O[1224 + o] = c222;   // (2,2,2)
      O[1736 + o] = c332;   // (3,3,2)
    }
  }
#undef M0_
#undef M1_
#undef M2F
#undef M3F
}

extern "C" void kernel_launch(void* const* d_in, const int* in_sizes, int n_in,
                              void* d_out, int out_size, void* d_ws, size_t ws_size,
                              hipStream_t stream) {
  const float* drv = (const float*)d_in[0];   // (400000, 3) f32
  const int*   Z   = (const int*)  d_in[1];   // (10000,)    i32
  const int*   nbr = (const int*)  d_in[2];   // (2, 400000) i32
  const float* W   = (const float*)d_in[3];   // (10,10,8,16) f32
  float* out = (float*)d_out;                 // (10000, 2248) f32

  char* ws = (char*)d_ws;
  float* mom   = (float*)ws;                                      // 6.4 MB
  int*  sorted = (int*)(ws + sizeof(float) * N_ATOMS * MOM_STRIDE);
  int*  ctrl   = sorted + MAX_SORT;                               // ~1.2 KB

  const int initN = N_ATOMS * MOM_STRIDE;     // largest region to init
  k_init   <<<(initN + 255) / 256, 256, 0, stream>>>(mom, sorted, ctrl);
  k_hist   <<<(N_EDGES + 255) / 256, 256, 0, stream>>>(Z, nbr, ctrl);
  k_scan   <<<1, 1, 0, stream>>>(ctrl);
  k_scatter<<<(N_EDGES + 255) / 256, 256, 0, stream>>>(Z, nbr, ctrl, sorted);

  const int maxTiles = (N_EDGES + NPAIR * 15 + 15) / 16;  // 25094
  k_edges  <<<(maxTiles + 7) / 8, 256, 0, stream>>>(drv, Z, nbr, W,
                                                    sorted, ctrl, mom);
  k_atoms  <<<N_ATOMS / 8, 256, 0, stream>>>(mom, out);
}
